// BitLinear_83382495084947
// MI455X (gfx1250) — compile-verified, matrix operand order checked by reading, not run on hardware
//
#include <hip/hip_runtime.h>

// ---------------------------------------------------------------------------
// Problem constants (from reference)
// ---------------------------------------------------------------------------
#define IN_F   2048      // K
#define OUT_F  2048      // N
#define M_TOT  16384     // 8 * 2048 rows of activation
#define NBITS  4

// GEMM tiling
#define BM     256       // block tile M
#define BN     128       // block tile N
#define BK     64        // K step: 128B per row -> full-cacheline staging loads
#define LDA    (BK + 8)  // padded LDS row stride (72 elems = 144B, 16B aligned)

typedef __bf16 bf16_t;
typedef __attribute__((ext_vector_type(16))) __bf16        v16bf;
typedef __attribute__((ext_vector_type(4)))  __bf16        v4bf;
typedef __attribute__((ext_vector_type(8)))  float         v8f;
typedef __attribute__((ext_vector_type(4)))  float         v4f;
typedef __attribute__((ext_vector_type(4)))  unsigned int  v4u;
typedef __attribute__((ext_vector_type(4)))  int           v4i_t;

// address-space qualified vector pointers for the async global->LDS builtin:
// param 1 is v4i in addrspace(1) ("__device__"), param 2 is the LDS side (AS3)
typedef __attribute__((address_space(1))) v4i_t* as1_v4i;
typedef __attribute__((address_space(3))) v4i_t* as3_v4i;

union Frag16 {          // 16 bf16 = 32 bytes = two 16-byte loads
    v16bf v;
    v4u   q[2];
};

#if defined(__HIP_DEVICE_COMPILE__) && __has_builtin(__builtin_amdgcn_global_load_async_to_lds_b128)
#define USE_ASYNC_LDS 1
#else
#define USE_ASYNC_LDS 0
#endif

__device__ __forceinline__ void wait_async_zero() {
#if __has_builtin(__builtin_amdgcn_s_wait_asynccnt)
    __builtin_amdgcn_s_wait_asynccnt(0);
#else
    asm volatile("s_wait_asynccnt 0x0" ::: "memory");
#endif
}

__device__ __forceinline__ float sigmoidf_fast(float x) {
    return 1.0f / (1.0f + __expf(-x));
}

// ---------------------------------------------------------------------------
// Kernel 1: activation f32 -> bf16 (memory-bound streaming convert)
// ---------------------------------------------------------------------------
__global__ __launch_bounds__(256)
void k_cvt_input(const float* __restrict__ in, bf16_t* __restrict__ out, int n4) {
    int i = blockIdx.x * blockDim.x + threadIdx.x;
    if (i >= n4) return;
    v4f f = ((const v4f*)in)[i];
    v4bf o;
    o.x = (__bf16)f.x; o.y = (__bf16)f.y; o.z = (__bf16)f.z; o.w = (__bf16)f.w;
    ((v4bf*)out)[i] = o;
}

// ---------------------------------------------------------------------------
// Kernel 2: build dense weight matrix W[o,i] in bf16
//   W = scale * sum_n (sigmoid(pw) - sigmoid(nw)) * exps[n] * sigmoid(mask[n])
// ---------------------------------------------------------------------------
__global__ __launch_bounds__(256)
void k_build_weight(const float* __restrict__ pw, const float* __restrict__ nw,
                    const float* __restrict__ exps, const float* __restrict__ maskw,
                    const float* __restrict__ scale, bf16_t* __restrict__ wbf, int nElems) {
    int idx = blockIdx.x * blockDim.x + threadIdx.x;  // one (o,i) pair
    if (idx >= nElems) return;
    const float s  = scale[0];
    const float c0 = exps[0] * sigmoidf_fast(maskw[0]) * s;
    const float c1 = exps[1] * sigmoidf_fast(maskw[1]) * s;
    const float c2 = exps[2] * sigmoidf_fast(maskw[2]) * s;
    const float c3 = exps[3] * sigmoidf_fast(maskw[3]) * s;

    v4f p = ((const v4f*)pw)[idx];   // 4 bit-planes, contiguous
    v4f q = ((const v4f*)nw)[idx];

    float w = c0 * (sigmoidf_fast(p.x) - sigmoidf_fast(q.x))
            + c1 * (sigmoidf_fast(p.y) - sigmoidf_fast(q.y))
            + c2 * (sigmoidf_fast(p.z) - sigmoidf_fast(q.z))
            + c3 * (sigmoidf_fast(p.w) - sigmoidf_fast(q.w));
    wbf[idx] = (__bf16)w;
}

// ---------------------------------------------------------------------------
// Kernel 3: quantized bias (bit_ste) -> f32
// ---------------------------------------------------------------------------
__global__ __launch_bounds__(256)
void k_build_bias(const float* __restrict__ pb, const float* __restrict__ nb,
                  const float* __restrict__ bexps, const float* __restrict__ bscale,
                  float* __restrict__ bias) {
    int o = blockIdx.x * blockDim.x + threadIdx.x;
    if (o >= OUT_F) return;
    float r = 0.0f;
#pragma unroll
    for (int n = 0; n < NBITS; ++n)
        r += (pb[o * NBITS + n] - nb[o * NBITS + n]) * bexps[n];
    r = fminf(1.0f, fmaxf(-1.0f, r));
    float a = fabsf(r);
    float qv = nearbyintf(a * 15.0f) * (1.0f / 15.0f);
    qv = (r < 0.0f) ? -qv : qv;
    bias[o] = qv * bscale[0];
}

// ---------------------------------------------------------------------------
// Kernel 4: WMMA bf16 GEMM with double-buffered LDS staging
//   y[m,n] = sum_k A[m,k] * W[n,k] + bias[n]
//   block = 256 threads = 8 waves (4 along M x 2 along N)
//   block tile 256x128, wave tile 64x64 (4x4 fragments), BK=64
//   staging: coalesced full-cacheline async global->LDS
// ---------------------------------------------------------------------------
__global__ __launch_bounds__(256)
void k_gemm_wmma(const bf16_t* __restrict__ A, const bf16_t* __restrict__ W,
                 const float* __restrict__ bias, float* __restrict__ out) {
    const int K = IN_F, N = OUT_F;

    __shared__ bf16_t As[2][BM * LDA];   // 2 * 256 * 72 * 2B = 73728 B
    __shared__ bf16_t Bs[2][BN * LDA];   // 2 * 128 * 72 * 2B = 36864 B

    const int tid  = threadIdx.x;
    const int lane = tid & 31;
    const int wave = tid >> 5;
    const int wm   = wave & 3;              // 4 waves along M
    const int wn   = wave >> 2;             // 2 waves along N
    const int mBlock = blockIdx.x * BM;
    const int nBlock = blockIdx.y * BN;

    const int l  = lane & 15;               // row within 16-tile
    const int hi = lane >> 4;               // half-wave selects K sub-chunk

    // one 16B chunk: global -> LDS
    auto copy16 = [&](const bf16_t* g, bf16_t* s) {
#if USE_ASYNC_LDS
        __builtin_amdgcn_global_load_async_to_lds_b128(
            (as1_v4i)(v4i_t*)g,             // strip const, then addrspacecast
            (as3_v4i)(v4i_t*)s, 0, 0);
#else
        *(v4u*)s = *(const v4u*)g;
#endif
    };

    // stage one BK slice of A (256 rows x 128B) and B (128 rows x 128B)
    auto stage = [&](int buf, int k0) {
#pragma unroll
        for (int r = 0; r < 8; ++r) {       // A: 2048 chunks / 256 threads
            int lin = r * 256 + tid;
            int row = lin >> 3;             // 0..255
            int col = (lin & 7) * 8;        // elem offset within BK
            copy16(A + (size_t)(mBlock + row) * K + k0 + col,
                   &As[buf][row * LDA + col]);
        }
#pragma unroll
        for (int r = 0; r < 4; ++r) {       // B: 1024 chunks / 256 threads
            int lin = r * 256 + tid;
            int row = lin >> 3;             // 0..127
            int col = (lin & 7) * 8;
            copy16(W + (size_t)(nBlock + row) * K + k0 + col,
                   &Bs[buf][row * LDA + col]);
        }
    };

    v8f acc[4][4];
#pragma unroll
    for (int i = 0; i < 4; ++i)
#pragma unroll
        for (int j = 0; j < 4; ++j)
            acc[i][j] = (v8f){0.f, 0.f, 0.f, 0.f, 0.f, 0.f, 0.f, 0.f};

    stage(0, 0);
#if USE_ASYNC_LDS
    wait_async_zero();
#endif
    __syncthreads();

    const int NT = K / BK;                  // 32 K-tiles
    for (int kt = 0; kt < NT; ++kt) {
        const int cur = kt & 1;
        if (kt + 1 < NT)
            stage(cur ^ 1, (kt + 1) * BK);  // prefetch next tile (async)

        // fragment base pointers in LDS (ISA 7.12.2 16-bit A/B layout)
        const bf16_t* aRow[4];
        const bf16_t* bRow[4];
#pragma unroll
        for (int i = 0; i < 4; ++i) {
            aRow[i] = &As[cur][(wm * 64 + i * 16 + l) * LDA + hi * 8];
            bRow[i] = &Bs[cur][(wn * 64 + i * 16 + l) * LDA + hi * 8];
        }

#pragma unroll
        for (int ks = 0; ks < 2; ++ks) {    // two 16x16x32 sub-steps
            Frag16 fa[4], fb[4];
#pragma unroll
            for (int i = 0; i < 4; ++i) {
                const bf16_t* pa = aRow[i] + ks * 32;
                fa[i].q[0] = *(const v4u*)pa;
                fa[i].q[1] = *(const v4u*)(pa + 16);
                const bf16_t* pb = bRow[i] + ks * 32;
                fb[i].q[0] = *(const v4u*)pb;
                fb[i].q[1] = *(const v4u*)(pb + 16);
            }
#pragma unroll
            for (int i = 0; i < 4; ++i)
#pragma unroll
                for (int j = 0; j < 4; ++j)
                    acc[i][j] = __builtin_amdgcn_wmma_f32_16x16x32_bf16(
                        false, fa[i].v, false, fb[j].v,
                        (short)0, acc[i][j], false, false);
        }

#if USE_ASYNC_LDS
        wait_async_zero();                  // staged tile landed in LDS
#endif
        __syncthreads();                    // all waves done reading 'cur'
    }

    // D layout: lane = n + 16*(m/8), VGPR v = m%8  (32-bit C/D 16x16)
    float bv[4];
#pragma unroll
    for (int j = 0; j < 4; ++j)
        bv[j] = bias[nBlock + wn * 64 + j * 16 + l];

#pragma unroll
    for (int i = 0; i < 4; ++i) {
        const int rowBase = mBlock + wm * 64 + i * 16 + hi * 8;
#pragma unroll
        for (int j = 0; j < 4; ++j) {
            const int col = nBlock + wn * 64 + j * 16 + l;
#pragma unroll
            for (int v = 0; v < 8; ++v)
                out[(size_t)(rowBase + v) * N + col] = acc[i][j][v] + bv[j];
        }
    }
}

// ---------------------------------------------------------------------------
// Launcher
// ---------------------------------------------------------------------------
extern "C" void kernel_launch(void* const* d_in, const int* in_sizes, int n_in,
                              void* d_out, int out_size, void* d_ws, size_t ws_size,
                              hipStream_t stream) {
    const float* input   = (const float*)d_in[0];  // [8,2048,2048]
    const float* pweight = (const float*)d_in[1];  // [O,I,4]
    const float* nweight = (const float*)d_in[2];  // [O,I,4]
    const float* exps    = (const float*)d_in[3];  // [4]
    const float* bexps   = (const float*)d_in[4];  // [4]
    const float* maskw   = (const float*)d_in[5];  // [4]
    const float* scale   = (const float*)d_in[6];  // [1]
    const float* pbias   = (const float*)d_in[7];  // [O,4]
    const float* nbias   = (const float*)d_in[8];  // [O,4]
    const float* bscale  = (const float*)d_in[9];  // [1]
    float* out = (float*)d_out;

    char* ws = (char*)d_ws;
    bf16_t* Abf  = (bf16_t*)ws;                                            // 67 MB
    bf16_t* Wbf  = (bf16_t*)(ws + (size_t)M_TOT * IN_F * sizeof(bf16_t));  //  8 MB
    float*  bias = (float*)(ws + (size_t)M_TOT * IN_F * sizeof(bf16_t)
                               + (size_t)OUT_F * IN_F * sizeof(bf16_t));   //  8 KB

    // 1) activation f32 -> bf16
    const int n4 = (M_TOT * IN_F) / 4;
    k_cvt_input<<<(n4 + 255) / 256, 256, 0, stream>>>(input, Abf, n4);

    // 2) dense weight build
    const int nw = OUT_F * IN_F;
    k_build_weight<<<(nw + 255) / 256, 256, 0, stream>>>(pweight, nweight, exps,
                                                         maskw, scale, Wbf, nw);

    // 3) quantized bias
    k_build_bias<<<(OUT_F + 255) / 256, 256, 0, stream>>>(pbias, nbias, bexps,
                                                          bscale, bias);

    // 4) WMMA GEMM: grid tiles 16384 x 2048 as (256 M x 128 N) blocks
    dim3 grid(M_TOT / BM, OUT_F / BN);
    k_gemm_wmma<<<grid, 256, 0, stream>>>(Abf, Wbf, bias, out);
}
